// GAT_1_53266184405175
// MI455X (gfx1250) — matrix operand order repaired and checked by reading, not compiled, not run on hardware
//
#include <hip/hip_runtime.h>
#include <hip/hip_bf16.h>
#include <math.h>

typedef __attribute__((ext_vector_type(16))) _Float16 v16h;
typedef __attribute__((ext_vector_type(8)))  _Float16 v8h;
typedef __attribute__((ext_vector_type(8)))  float    v8f;

#define NEG_SLOPE 0.2f
#define EPS_DENOM 1e-16f

__device__ __forceinline__ void atomicMaxF(float* addr, float val) {
    // Works for mixed signs when the slot is initialized to -inf (0xFF800000).
    if (val >= 0.0f) atomicMax((int*)addr, __float_as_int(val));
    else             atomicMin((unsigned int*)addr, __float_as_uint(val));
}

// ---------------------------------------------------------------- fills
__global__ void fill_kernel(float* __restrict__ p, float v, size_t n) {
    size_t i = (size_t)blockIdx.x * blockDim.x + threadIdx.x;
    if (i < n) p[i] = v;
}

// ---------------------------------------------------------------- f16 staging
// xh[N,128] = (f16) x[N,128]
__global__ void cvt_x_kernel(const float* __restrict__ x, _Float16* __restrict__ xh, size_t n) {
    size_t i = (size_t)blockIdx.x * blockDim.x + threadIdx.x;
    if (i < n) xh[i] = (_Float16)x[i];
}
// wt[C,K] = (f16) W[K,C]   (transpose so K is contiguous per output column)
__global__ void cvt_wT_kernel(const float* __restrict__ W, _Float16* __restrict__ wt,
                              int K, int C) {
    int i = blockIdx.x * blockDim.x + threadIdx.x;
    if (i >= K * C) return;
    int c = i / K, k = i % K;
    wt[(size_t)c * K + k] = (_Float16)W[(size_t)k * C + c];
}

// ---------------------------------------------------------------- GEMM1
// h1[N,64] = x[N,128] @ W1[128,64]   (v_wmma_f32_16x16x32_f16)
// One wave -> one 16-row slab, all 64 output columns (4 tiles), K-loop of 4.
// A from xh (f16, row-major), B from w1t (f16, [64 cols][128 K]).
__global__ void gemm1_wmma_kernel(const _Float16* __restrict__ xh,
                                  const _Float16* __restrict__ w1t,
                                  float* __restrict__ h1, int N) {
    int gwave = (int)(((size_t)blockIdx.x * blockDim.x + threadIdx.x) >> 5);
    int lane  = threadIdx.x & 31;
    int r0    = gwave * 16;
    if (r0 + 16 > N) return;                 // wave-uniform guard (EXEC stays full)
    int m = lane & 15, g = lane >> 4;

    v8f acc[4] = {};
    for (int kt = 0; kt < 4; ++kt) {
        int K0 = kt * 32;
        // A fragment: 16x32 f16. Lane-group g holds K = {K0+8g..+7} and {+16}.
        const _Float16* arow = xh + (size_t)(r0 + m) * 128 + K0 + 8 * g;
        if (kt < 3) __builtin_prefetch(arow + 32, 0, 3);   // global_prefetch_b8
        v8h alo = *(const v8h*)(arow);
        v8h ahi = *(const v8h*)(arow + 16);
        v16h a;
#pragma unroll
        for (int j = 0; j < 8; ++j) { a[j] = alo[j]; a[8 + j] = ahi[j]; }

#pragma unroll
        for (int t = 0; t < 4; ++t) {
            // B fragment: 32x16 f16. Lane-group g holds K = {K0+8g..+7} and {+16}, col = m.
            const _Float16* brow = w1t + (size_t)(t * 16 + m) * 128 + K0 + 8 * g;
            v8h blo = *(const v8h*)(brow);
            v8h bhi = *(const v8h*)(brow + 16);
            v16h b;
#pragma unroll
            for (int j = 0; j < 8; ++j) { b[j] = blo[j]; b[8 + j] = bhi[j]; }
            acc[t] = __builtin_amdgcn_wmma_f32_16x16x32_f16(
                false, a, false, b, (short)0, acc[t], false, false);
        }
    }
#pragma unroll
    for (int t = 0; t < 4; ++t)
#pragma unroll
        for (int v = 0; v < 8; ++v)
            h1[(size_t)(r0 + v + 8 * g) * 64 + t * 16 + m] = acc[t][v];
}

// ---------------------------------------------------------------- GEMM2
// t2[N,16] = h2f16[N,64] @ W2[64,16]   (B from w2t f16 [16 cols][64 K])
__global__ void gemm2_wmma_kernel(const _Float16* __restrict__ h2,
                                  const _Float16* __restrict__ w2t,
                                  float* __restrict__ t2, int N) {
    int gwave = (int)(((size_t)blockIdx.x * blockDim.x + threadIdx.x) >> 5);
    int lane  = threadIdx.x & 31;
    int r0    = gwave * 16;
    if (r0 + 16 > N) return;
    int m = lane & 15, g = lane >> 4;

    v8f acc = {};
#pragma unroll
    for (int kt = 0; kt < 2; ++kt) {
        int K0 = kt * 32;
        const _Float16* arow = h2 + (size_t)(r0 + m) * 64 + K0 + 8 * g;
        v8h alo = *(const v8h*)(arow);
        v8h ahi = *(const v8h*)(arow + 16);
        v16h a;
#pragma unroll
        for (int j = 0; j < 8; ++j) { a[j] = alo[j]; a[8 + j] = ahi[j]; }
        const _Float16* brow = w2t + (size_t)m * 64 + K0 + 8 * g;
        v8h blo = *(const v8h*)(brow);
        v8h bhi = *(const v8h*)(brow + 16);
        v16h b;
#pragma unroll
        for (int j = 0; j < 8; ++j) { b[j] = blo[j]; b[8 + j] = bhi[j]; }
        acc = __builtin_amdgcn_wmma_f32_16x16x32_f16(
            false, a, false, b, (short)0, acc, false, false);
    }
#pragma unroll
    for (int v = 0; v < 8; ++v)
        t2[(size_t)(r0 + v + 8 * g) * 16 + m] = acc[v];
}

// ---------------------------------------------------------------- attention dots
__global__ void attn1_kernel(const float* __restrict__ h1,
                             const float* __restrict__ att_s,
                             const float* __restrict__ att_d,
                             float* __restrict__ as, float* __restrict__ ad, int N) {
    int idx = blockIdx.x * blockDim.x + threadIdx.x;
    if (idx >= N * 8) return;
    int n = idx >> 3, h = idx & 7;
    float ss = 0.f, dd = 0.f;
#pragma unroll
    for (int c = 0; c < 8; ++c) {
        float v = h1[(size_t)n * 64 + h * 8 + c];
        ss += v * att_s[h * 8 + c];
        dd += v * att_d[h * 8 + c];
    }
    as[idx] = ss; ad[idx] = dd;
}

__global__ void attn2_kernel(const float* __restrict__ t2,
                             const float* __restrict__ att_s,
                             const float* __restrict__ att_d,
                             float* __restrict__ as, float* __restrict__ ad, int N) {
    int n = blockIdx.x * blockDim.x + threadIdx.x;
    if (n >= N) return;
    float ss = 0.f, dd = 0.f;
#pragma unroll
    for (int c = 0; c < 16; ++c) {
        float v = t2[(size_t)n * 16 + c];
        ss += v * att_s[c];
        dd += v * att_d[c];
    }
    as[n] = ss; ad[n] = dd;
}

// ---------------------------------------------------------------- edge helpers
__device__ __forceinline__ void edge_sd(const long long* ei, size_t e, int E, int* s, int* d) {
    if (e < (size_t)E) { *s = (int)ei[e]; *d = (int)ei[(size_t)E + e]; }
    else               { *s = *d = (int)(e - (size_t)E); }
}

// ---------------------------------------------------------------- layer-1 edge passes (H=8)
__global__ void edge1_max_kernel(const long long* __restrict__ ei, int E, int N,
                                 const float* __restrict__ as, const float* __restrict__ ad,
                                 float* __restrict__ alpha, float* __restrict__ amax) {
    size_t e = (size_t)blockIdx.x * blockDim.x + threadIdx.x;
    if (e >= (size_t)E + N) return;
    int s, d; edge_sd(ei, e, E, &s, &d);
#pragma unroll
    for (int h = 0; h < 8; ++h) {
        float a = as[(size_t)s * 8 + h] + ad[(size_t)d * 8 + h];
        a = a > 0.0f ? a : NEG_SLOPE * a;          // leaky_relu
        alpha[e * 8 + h] = a;
        atomicMaxF(&amax[(size_t)d * 8 + h], a);
    }
}

__global__ void edge1_sum_kernel(const long long* __restrict__ ei, int E, int N,
                                 float* __restrict__ alpha,
                                 const float* __restrict__ amax, float* __restrict__ denom) {
    size_t e = (size_t)blockIdx.x * blockDim.x + threadIdx.x;
    if (e >= (size_t)E + N) return;
    int s, d; edge_sd(ei, e, E, &s, &d); (void)s;
#pragma unroll
    for (int h = 0; h < 8; ++h) {
        float mv = amax[(size_t)d * 8 + h];
        if (!isfinite(mv)) mv = 0.0f;
        float ex = __expf(alpha[e * 8 + h] - mv);
        alpha[e * 8 + h] = ex;                     // keep exp for pass 3
        atomicAdd(&denom[(size_t)d * 8 + h], ex);
    }
}

__global__ void edge1_agg_kernel(const long long* __restrict__ ei, int E, int N,
                                 const float* __restrict__ alpha, const float* __restrict__ denom,
                                 const float* __restrict__ h1, float* __restrict__ out1) {
    size_t idx = (size_t)blockIdx.x * blockDim.x + threadIdx.x;
    size_t total = ((size_t)E + N) * 64;
    if (idx >= total) return;
    size_t e = idx >> 6; int k = (int)(idx & 63); int h = k >> 3;
    int s, d; edge_sd(ei, e, E, &s, &d);
    float w = alpha[e * 8 + h] / (denom[(size_t)d * 8 + h] + EPS_DENOM);
    atomicAdd(&out1[(size_t)d * 64 + k], w * h1[(size_t)s * 64 + k]);
}

// ---------------------------------------------------------------- ELU + f16 pack
__global__ void h2_elu_kernel(const float* __restrict__ out1, const float* __restrict__ b1,
                              _Float16* __restrict__ h2, size_t n) {
    size_t idx = (size_t)blockIdx.x * blockDim.x + threadIdx.x;
    if (idx >= n) return;
    float v = out1[idx] + b1[idx & 63];
    v = v > 0.0f ? v : (__expf(v) - 1.0f);         // elu
    h2[idx] = (_Float16)v;
}

// ---------------------------------------------------------------- layer-2 edge passes (H=1)
__global__ void edge2_max_kernel(const long long* __restrict__ ei, int E, int N,
                                 const float* __restrict__ as, const float* __restrict__ ad,
                                 float* __restrict__ alpha, float* __restrict__ amax) {
    size_t e = (size_t)blockIdx.x * blockDim.x + threadIdx.x;
    if (e >= (size_t)E + N) return;
    int s, d; edge_sd(ei, e, E, &s, &d);
    float a = as[s] + ad[d];
    a = a > 0.0f ? a : NEG_SLOPE * a;
    alpha[e] = a;
    atomicMaxF(&amax[d], a);
}

__global__ void edge2_sum_kernel(const long long* __restrict__ ei, int E, int N,
                                 float* __restrict__ alpha,
                                 const float* __restrict__ amax, float* __restrict__ denom) {
    size_t e = (size_t)blockIdx.x * blockDim.x + threadIdx.x;
    if (e >= (size_t)E + N) return;
    int s, d; edge_sd(ei, e, E, &s, &d); (void)s;
    float mv = amax[d];
    if (!isfinite(mv)) mv = 0.0f;
    float ex = __expf(alpha[e] - mv);
    alpha[e] = ex;
    atomicAdd(&denom[d], ex);
}

__global__ void edge2_agg_kernel(const long long* __restrict__ ei, int E, int N,
                                 const float* __restrict__ alpha, const float* __restrict__ denom,
                                 const float* __restrict__ t2, float* __restrict__ out2) {
    size_t idx = (size_t)blockIdx.x * blockDim.x + threadIdx.x;
    size_t total = ((size_t)E + N) * 16;
    if (idx >= total) return;
    size_t e = idx >> 4; int c = (int)(idx & 15);
    int s, d; edge_sd(ei, e, E, &s, &d);
    float w = alpha[e] / (denom[d] + EPS_DENOM);
    atomicAdd(&out2[(size_t)d * 16 + c], w * t2[(size_t)s * 16 + c]);
}

// ---------------------------------------------------------------- bias + log_softmax
__global__ void lsm_kernel(const float* __restrict__ out2, const float* __restrict__ b2,
                           float* __restrict__ y, int N) {
    int n = blockIdx.x * blockDim.x + threadIdx.x;
    if (n >= N) return;
    float z[16], mx = -INFINITY;
#pragma unroll
    for (int c = 0; c < 16; ++c) { z[c] = out2[(size_t)n * 16 + c] + b2[c]; mx = fmaxf(mx, z[c]); }
    float s = 0.f;
#pragma unroll
    for (int c = 0; c < 16; ++c) s += __expf(z[c] - mx);
    float lse = __logf(s) + mx;
#pragma unroll
    for (int c = 0; c < 16; ++c) y[(size_t)n * 16 + c] = z[c] - lse;
}

// ---------------------------------------------------------------- launch
static inline unsigned cdiv(size_t a, size_t b) { return (unsigned)((a + b - 1) / b); }

extern "C" void kernel_launch(void* const* d_in, const int* in_sizes, int n_in,
                              void* d_out, int out_size, void* d_ws, size_t ws_size,
                              hipStream_t stream) {
    const float*     x        = (const float*)d_in[0];
    const long long* ei       = (const long long*)d_in[1];   // int64 [2,E]
    const float*     W1       = (const float*)d_in[2];
    const float*     b1       = (const float*)d_in[3];
    const float*     att_src1 = (const float*)d_in[4];
    const float*     att_dst1 = (const float*)d_in[5];
    const float*     W2       = (const float*)d_in[6];
    const float*     b2       = (const float*)d_in[7];
    const float*     att_src2 = (const float*)d_in[8];
    const float*     att_dst2 = (const float*)d_in[9];

    int N  = in_sizes[0] / 128;
    int E  = in_sizes[1] / 2;
    size_t Ep = (size_t)E + N;

    // workspace carve-out (256B aligned regions)
    size_t off = 0;
    auto carve = [&](size_t bytes) -> void* {
        void* p = (char*)d_ws + off;
        off += (bytes + 255) & ~(size_t)255;
        return p;
    };
    float*     h1     = (float*)carve((size_t)N * 64 * 4);
    float*     as1    = (float*)carve((size_t)N * 8 * 4);
    float*     ad1    = (float*)carve((size_t)N * 8 * 4);
    float*     amax1  = (float*)carve((size_t)N * 8 * 4);
    float*     den1   = (float*)carve((size_t)N * 8 * 4);
    float*     out1   = (float*)carve((size_t)N * 64 * 4);
    float*     alpha  = (float*)carve(Ep * 8 * 4);           // reused for layer 2 (needs Ep*4)
    _Float16*  h2     = (_Float16*)carve((size_t)N * 64 * 2);
    float*     t2     = (float*)carve((size_t)N * 16 * 4);
    float*     as2    = (float*)carve((size_t)N * 4);
    float*     ad2    = (float*)carve((size_t)N * 4);
    float*     amax2  = (float*)carve((size_t)N * 4);
    float*     den2   = (float*)carve((size_t)N * 4);
    float*     out2   = (float*)carve((size_t)N * 16 * 4);
    _Float16*  xh     = (_Float16*)carve((size_t)N * 128 * 2);
    _Float16*  w1t    = (_Float16*)carve((size_t)128 * 64 * 2);
    _Float16*  w2t    = (_Float16*)carve((size_t)64 * 16 * 2);
    (void)ws_size; (void)n_in; (void)out_size;

    const int T = 256;

    // init accumulators + f16 staging
    fill_kernel<<<cdiv((size_t)N * 8, T), T, 0, stream>>>(amax1, -INFINITY, (size_t)N * 8);
    fill_kernel<<<cdiv((size_t)N * 8, T), T, 0, stream>>>(den1, 0.0f, (size_t)N * 8);
    fill_kernel<<<cdiv((size_t)N * 64, T), T, 0, stream>>>(out1, 0.0f, (size_t)N * 64);
    fill_kernel<<<cdiv((size_t)N, T), T, 0, stream>>>(amax2, -INFINITY, (size_t)N);
    fill_kernel<<<cdiv((size_t)N, T), T, 0, stream>>>(den2, 0.0f, (size_t)N);
    fill_kernel<<<cdiv((size_t)N * 16, T), T, 0, stream>>>(out2, 0.0f, (size_t)N * 16);
    cvt_x_kernel<<<cdiv((size_t)N * 128, T), T, 0, stream>>>(x, xh, (size_t)N * 128);
    cvt_wT_kernel<<<cdiv((size_t)128 * 64, T), T, 0, stream>>>(W1, w1t, 128, 64);
    cvt_wT_kernel<<<cdiv((size_t)64 * 16, T), T, 0, stream>>>(W2, w2t, 64, 16);

    // ---- layer 1 ----
    {   // 1 wave per 16-row slab, 8 waves per block
        size_t threads = (size_t)(N / 16) * 32;
        gemm1_wmma_kernel<<<cdiv(threads, T), T, 0, stream>>>(xh, w1t, h1, N);
    }
    attn1_kernel<<<cdiv((size_t)N * 8, T), T, 0, stream>>>(h1, att_src1, att_dst1, as1, ad1, N);
    edge1_max_kernel<<<cdiv(Ep, T), T, 0, stream>>>(ei, E, N, as1, ad1, alpha, amax1);
    edge1_sum_kernel<<<cdiv(Ep, T), T, 0, stream>>>(ei, E, N, alpha, amax1, den1);
    edge1_agg_kernel<<<cdiv(Ep * 64, T), T, 0, stream>>>(ei, E, N, alpha, den1, h1, out1);
    h2_elu_kernel<<<cdiv((size_t)N * 64, T), T, 0, stream>>>(out1, b1, h2, (size_t)N * 64);

    // ---- layer 2 ----
    {
        size_t threads = (size_t)(N / 16) * 32;
        gemm2_wmma_kernel<<<cdiv(threads, T), T, 0, stream>>>(h2, w2t, t2, N);
    }
    attn2_kernel<<<cdiv((size_t)N, T), T, 0, stream>>>(t2, att_src2, att_dst2, as2, ad2, N);
    edge2_max_kernel<<<cdiv(Ep, T), T, 0, stream>>>(ei, E, N, as2, ad2, alpha, amax2);
    edge2_sum_kernel<<<cdiv(Ep, T), T, 0, stream>>>(ei, E, N, alpha, amax2, den2);
    edge2_agg_kernel<<<cdiv(Ep * 16, T), T, 0, stream>>>(ei, E, N, alpha, den2, t2, out2);

    lsm_kernel<<<cdiv((size_t)N, T), T, 0, stream>>>(out2, b2, (float*)d_out, N);
}
